// UnigramProcessor_70291434766372
// MI455X (gfx1250) — matrix operand order repaired and checked by reading, not compiled, not run on hardware
//
#include <hip/hip_runtime.h>
#include <hip/hip_bf16.h>
#include <math.h>
#include <stdint.h>

// CDNA5 / gfx1250 implementation of the unigram-watermark top-p sampler.
//
//   K1: online softmax stats per row (max, logsumexp)        - streams scores once
//   K2: 256-bin log-prob histogram in LDS + WMMA prefix scan - L2-resident re-read
//   K3: Gumbel-max sampling over kept tokens, with TDM
//       (tensor_load_to_lds) double-buffered row tiles       - L2-resident re-read
//   K4: nontemporal float4 fill 1e-5 + 1e5 at sampled token  - the mandatory write
//
// 192 MB L2 holds the 131 MB scores tensor after K1, so K2/K3 run at L2 BW.

typedef __attribute__((ext_vector_type(16))) _Float16 v16h;
typedef __attribute__((ext_vector_type(8)))  float    v8f;
typedef __attribute__((ext_vector_type(4)))  float    v4f;
typedef __attribute__((ext_vector_type(4)))  unsigned int v4u;
typedef __attribute__((ext_vector_type(8)))  int      v8i;
typedef __attribute__((ext_vector_type(4)))  int      v4i;

#define BATCH   256
#define VOCAB   128000
#define BIASF   2.0f
#define TOPP    0.9f
#define NBINS   256
#define LRANGE  32.0f       // histogram covers logp in [-32, 0]
#define NEGBIG  -3.4e38f

#define TILE_ELEMS 6400                 // 25.6 KB tiles; 20 tiles cover VOCAB exactly
#define NTILES     (VOCAB / TILE_ELEMS)

#if __has_builtin(__builtin_amdgcn_tensor_load_to_lds) && \
    __has_builtin(__builtin_amdgcn_s_wait_tensorcnt)
#define USE_TDM 1
#else
#define USE_TDM 0
#endif

// ---------------------------------------------------------------- utilities

__device__ __forceinline__ float u01_hash(uint32_t b, uint32_t v) {
    // splitmix64-style deterministic hash -> uniform (0,1)
    uint64_t z = (((uint64_t)b << 32) | v) + 0x9E3779B97F4A7C15ULL;
    z = (z ^ (z >> 30)) * 0xBF58476D1CE4E5B9ULL;
    z = (z ^ (z >> 27)) * 0x94D049BB133111EBULL;
    z ^= z >> 31;
    uint32_t r = (uint32_t)(z >> 32);
    return ((float)r + 0.5f) * (1.0f / 4294967296.0f);
}

#if USE_TDM
// Issue a TDM DMA of one 1-row tile (TILE_ELEMS f32) global -> LDS.
// D# packing per CDNA5 ISA ch.8: group0 = {ctrl, lds_addr, global_addr, type=2},
// group1 = {data_size=4B, tensor_dim0/1, tile_dim0/1, dim0 stride}.
__device__ __forceinline__ void tdm_issue_tile(void* lds_dst, const float* gsrc) {
    unsigned lds_off = (unsigned)(unsigned long long)(uintptr_t)lds_dst; // LDS aperture low 32b
    unsigned long long ga = (unsigned long long)(uintptr_t)gsrc;

    v4u g0;
    g0.x = 0x1u;                                   // count=1, user descriptor
    g0.y = lds_off;                                // lds_addr (bytes)
    g0.z = (unsigned)(ga & 0xFFFFFFFFu);           // global_addr[31:0]
    g0.w = (unsigned)((ga >> 32) & 0x01FFFFFFu)    // global_addr[56:32]
         | (2u << 30);                             // type = 2 ("image")

    v8i g1;
    g1[0] = (int)(2u << 16);                       // wg_mask=0, data_size=2 (4 bytes)
    g1[1] = (int)(((unsigned)TILE_ELEMS & 0xFFFFu) << 16);   // tensor_dim0[15:0] @63:48
    g1[2] = (int)(((unsigned)TILE_ELEMS >> 16) | (1u << 16)); // tensor_dim0[31:16], tensor_dim1=1
    g1[3] = (int)((unsigned)TILE_ELEMS << 16);     // tensor_dim1[31:16]=0, tile_dim0
    g1[4] = 1;                                     // tile_dim1=1, tile_dim2=0 (unused)
    g1[5] = (int)TILE_ELEMS;                       // tensor_dim0_stride[31:0]
    g1[6] = 0;                                     // stride0[47:32]=0, stride1[15:0]=0
    g1[7] = 0;                                     // stride1[47:16]=0

    v4i g2 = {0, 0, 0, 0};                         // dims 2..4 unused
    v4i g3 = {0, 0, 0, 0};

#if defined(__clang_major__) && (__clang_major__ >= 23)
    v8i g4 = {0, 0, 0, 0, 0, 0, 0, 0};
    __builtin_amdgcn_tensor_load_to_lds(g0, g1, g2, g3, g4, 0);
#else
    __builtin_amdgcn_tensor_load_to_lds(g0, g1, g2, g3, 0);
#endif
}
#endif // USE_TDM

// ------------------------------------------------------- K1: softmax stats

__global__ __launch_bounds__(512) void k_softmax_stats(
    const float* __restrict__ scores, const float* __restrict__ green,
    float* __restrict__ rowmax, float* __restrict__ rowlogsum)
{
    const int b = blockIdx.x;
    const float4* sp = (const float4*)(scores + (size_t)b * VOCAB);
    const float4* gp = (const float4*)green;
    const int n4 = VOCAB / 4;

    float m = NEGBIG, s = 0.f;
    for (int i = threadIdx.x; i < n4; i += blockDim.x) {
        __builtin_prefetch(&sp[i + 4 * 512], 0, 0);   // global_prefetch_b8
        float4 sv = sp[i];
        float4 gv = gp[i];
        float xs[4] = { sv.x + BIASF * gv.x, sv.y + BIASF * gv.y,
                        sv.z + BIASF * gv.z, sv.w + BIASF * gv.w };
        #pragma unroll
        for (int j = 0; j < 4; ++j) {
            float x = xs[j];
            if (x > m) { s = s * __expf(m - x) + 1.f; m = x; }
            else       { s += __expf(x - m); }
        }
    }
    #pragma unroll
    for (int off = 16; off > 0; off >>= 1) {
        float m2 = __shfl_xor(m, off, 32);
        float s2 = __shfl_xor(s, off, 32);
        float mn = fmaxf(m, m2);
        s = s * __expf(m - mn) + s2 * __expf(m2 - mn);
        m = mn;
    }
    __shared__ float sm[16], ss[16];
    const int lane = threadIdx.x & 31, wave = threadIdx.x >> 5;
    if (lane == 0) { sm[wave] = m; ss[wave] = s; }
    __syncthreads();
    if (threadIdx.x < 32) {
        m = (lane < 16) ? sm[lane] : NEGBIG;
        s = (lane < 16) ? ss[lane] : 0.f;
        #pragma unroll
        for (int off = 16; off > 0; off >>= 1) {
            float m2 = __shfl_xor(m, off, 32);
            float s2 = __shfl_xor(s, off, 32);
            float mn = fmaxf(m, m2);
            s = s * __expf(m - mn) + s2 * __expf(m2 - mn);
            m = mn;
        }
        if (lane == 0) { rowmax[b] = m; rowlogsum[b] = __logf(s); }
    }
}

// --------------------------------------- K2: histogram + WMMA scan cutoff

__global__ __launch_bounds__(512) void k_topp_cutoff(
    const float* __restrict__ scores, const float* __restrict__ green,
    const float* __restrict__ rowmax, const float* __restrict__ rowlogsum,
    float* __restrict__ cut)
{
    const int b = blockIdx.x;
    __shared__ float hist[NBINS];
    __shared__ float Tshare[16];

    for (int i = threadIdx.x; i < NBINS; i += blockDim.x) hist[i] = 0.f;
    __syncthreads();

    const float M = rowmax[b], LS = rowlogsum[b];
    const float4* sp = (const float4*)(scores + (size_t)b * VOCAB);
    const float4* gp = (const float4*)green;

    for (int i = threadIdx.x; i < VOCAB / 4; i += blockDim.x) {
        float4 sv = sp[i];
        float4 gv = gp[i];
        float xs[4] = { sv.x + BIASF * gv.x, sv.y + BIASF * gv.y,
                        sv.z + BIASF * gv.z, sv.w + BIASF * gv.w };
        #pragma unroll
        for (int j = 0; j < 4; ++j) {
            float t = xs[j] - M - LS;               // log-prob, <= 0
            float p = __expf(t);                    // normalized probability
            int bin = (int)floorf((t + LRANGE) * ((float)NBINS / LRANGE));
            bin = min(max(bin, 0), NBINS - 1);
            atomicAdd(&hist[bin], p);               // ds_add_f32
        }
    }
    __syncthreads();

    // Prefix scan of the reversed histogram via one WMMA:
    // Hrev (16x16 f16, zero-padded to K=32) x upper-triangular ones.
    const int lane = threadIdx.x;
    v8f c = {};
    if (threadIdx.x < 32) {
        const bool hi = lane >= 16;
        const int rowA = lane & 15;
        v16h a = {};
        #pragma unroll
        for (int i = 0; i < 8; ++i) {               // A halfs 0..7 = K 0..7 (lo) / 8..15 (hi)
            int k = i + (hi ? 8 : 0);
            a[i] = (_Float16)hist[255 - (16 * rowA + k)];
        }
        const int colB = lane & 15;
        v16h bu = {};
        #pragma unroll
        for (int i = 0; i < 16; ++i) {              // B half i = K 8h+i (i<8) / 16+8h+(i-8) (i>=8)
            int k = (i < 8) ? (i + (hi ? 8 : 0)) : (16 + (hi ? 8 : 0) + (i - 8));
            bu[i] = (_Float16)((k <= colB) ? 1.0f : 0.0f);
        }
        c = __builtin_amdgcn_wmma_f32_16x16x32_f16(
                false, a, false, bu, (short)0, c, false, false);
        if ((lane & 15) == 15) {                    // column 15 holds row totals
            #pragma unroll
            for (int i = 0; i < 8; ++i) Tshare[i + (hi ? 8 : 0)] = c[i];
        }
    }
    __syncthreads();
    if (threadIdx.x < 32) {
        const bool hi = lane >= 16;
        const int col = lane & 15;
        int best = 1 << 30;                          // first reversed-bin with cum >= TOPP
        #pragma unroll
        for (int i = 0; i < 8; ++i) {
            int row = i + (hi ? 8 : 0);
            float o = 0.f;
            for (int r = 0; r < row; ++r) o += Tshare[r];
            float cum = c[i] + o;
            int g = 16 * row + col;
            if (cum >= TOPP && g < best) best = g;
        }
        #pragma unroll
        for (int off = 16; off > 0; off >>= 1)
            best = min(best, __shfl_xor(best, off, 32));
        if (lane == 0) {
            if (best > NBINS - 1) best = NBINS - 1;  // fallback: keep everything
            int borig = (NBINS - 1) - best;          // original-bin index of crossing
            float tlow = (float)borig * (LRANGE / (float)NBINS) - LRANGE;
            cut[b] = M + LS + tlow;                  // absolute logit threshold
        }
    }
}

// --------------- K3: Gumbel-max over kept set, TDM double-buffered tiles

__global__ __launch_bounds__(256) void k_sample(
    const float* __restrict__ scores, const float* __restrict__ green,
    const float* __restrict__ cut, int* __restrict__ tok)
{
    const int b = blockIdx.x;
    const float cth = cut[b];
    const float* rowbase = scores + (size_t)b * VOCAB;

#if USE_TDM
    __shared__ float tile[2][TILE_ELEMS];
    if (threadIdx.x < 32) tdm_issue_tile(&tile[0][0], rowbase);   // prologue: tile 0
#endif

    float bv = NEGBIG; int bi = 0;
    for (int t = 0; t < NTILES; ++t) {
#if USE_TDM
        if (threadIdx.x < 32) {                      // wave 0 drives the DMA pipeline
            if (t + 1 < NTILES) {
                tdm_issue_tile(&tile[(t + 1) & 1][0],
                               rowbase + (size_t)(t + 1) * TILE_ELEMS);
                __builtin_amdgcn_s_wait_tensorcnt(1);   // tile t complete (in-order)
            } else {
                __builtin_amdgcn_s_wait_tensorcnt(0);   // drain
            }
        }
        __syncthreads();                             // publish tile t to all waves
        const float4* tp = (const float4*)&tile[t & 1][0];
#else
        const float4* tp = (const float4*)(rowbase + (size_t)t * TILE_ELEMS);
#endif
        const float4* gp = (const float4*)(green + (size_t)t * TILE_ELEMS);
        for (int j = threadIdx.x; j < TILE_ELEMS / 4; j += blockDim.x) {
            float4 sv = tp[j];
            float4 gv = gp[j];
            float xs[4] = { sv.x + BIASF * gv.x, sv.y + BIASF * gv.y,
                            sv.z + BIASF * gv.z, sv.w + BIASF * gv.w };
            #pragma unroll
            for (int jj = 0; jj < 4; ++jj) {
                float x = xs[jj];
                if (x >= cth) {
                    int v = t * TILE_ELEMS + 4 * j + jj;
                    float u = u01_hash((uint32_t)b, (uint32_t)v);
                    float g = -__logf(-__logf(u));   // Gumbel(0,1)
                    float key = x + g;               // argmax == truncated categorical
                    if (key > bv || (key == bv && v < bi)) { bv = key; bi = v; }
                }
            }
        }
#if USE_TDM
        __syncthreads();                             // tile t consumed before its buffer is reused
#endif
    }

    #pragma unroll
    for (int off = 16; off > 0; off >>= 1) {
        float ov = __shfl_xor(bv, off, 32);
        int   oi = __shfl_xor(bi, off, 32);
        if (ov > bv || (ov == bv && oi < bi)) { bv = ov; bi = oi; }
    }
    __shared__ float svs[8]; __shared__ int sis[8];
    const int lane = threadIdx.x & 31, wave = threadIdx.x >> 5;
    if (lane == 0) { svs[wave] = bv; sis[wave] = bi; }
    __syncthreads();
    if (threadIdx.x < 32) {
        bv = (lane < 8) ? svs[lane] : NEGBIG;
        bi = (lane < 8) ? sis[lane] : 0;
        #pragma unroll
        for (int off = 16; off > 0; off >>= 1) {
            float ov = __shfl_xor(bv, off, 32);
            int   oi = __shfl_xor(bi, off, 32);
            if (ov > bv || (ov == bv && oi < bi)) { bv = ov; bi = oi; }
        }
        if (lane == 0) tok[b] = bi;
    }
}

// ------------------------------------------------------- K4: output fill

__global__ __launch_bounds__(256) void k_fill(
    float* __restrict__ out, const int* __restrict__ tok)
{
    const int b = blockIdx.y;
    const int t = tok[b];                            // uniform scalar load
    const int i4 = blockIdx.x * blockDim.x + threadIdx.x;
    const int v = i4 * 4;
    if (v >= VOCAB) return;
    v4f val = { 1e-5f, 1e-5f, 1e-5f, 1e-5f };
    if (t >= v && t < v + 4) val[t - v] = 1e5f;
    // Nontemporal: don't evict the L2-resident scores with the 131MB result.
    __builtin_nontemporal_store(val, (v4f*)(out + (size_t)b * VOCAB) + i4);
}

// -------------------------------------------------------------- launcher

extern "C" void kernel_launch(void* const* d_in, const int* in_sizes, int n_in,
                              void* d_out, int out_size, void* d_ws, size_t ws_size,
                              hipStream_t stream) {
    (void)in_sizes; (void)n_in; (void)out_size; (void)ws_size;
    // setup_inputs order: input_ids (unused by the reference math), scores, green
    const float* scores = (const float*)d_in[1];
    const float* green  = (const float*)d_in[2];
    float* out = (float*)d_out;

    char*  ws      = (char*)d_ws;
    float* rowmax  = (float*)(ws + 0);
    float* rowls   = (float*)(ws + 1024);
    float* cut     = (float*)(ws + 2048);
    int*   tok     = (int*)  (ws + 3072);

    k_softmax_stats<<<BATCH, 512, 0, stream>>>(scores, green, rowmax, rowls);
    k_topp_cutoff  <<<BATCH, 512, 0, stream>>>(scores, green, rowmax, rowls, cut);
    k_sample       <<<BATCH, 256, 0, stream>>>(scores, green, cut, tok);

    dim3 g4((VOCAB / 4 + 255) / 256, BATCH);
    k_fill<<<g4, 256, 0, stream>>>(out, tok);
}
